// CoulombQeq_69441031242390
// MI455X (gfx1250) — matrix operand order repaired and checked by reading, not compiled, not run on hardware
//
#include <hip/hip_runtime.h>
#include <hip/hip_bf16.h>
#include <math.h>

typedef __attribute__((ext_vector_type(16))) _Float16 v16h;
typedef __attribute__((ext_vector_type(8)))  float    v8f;

#define B_  64
#define N_  512
#define D_  128
#define U_  64

#define KCOUL    14.399645351950548f
#define SQRT2F   1.41421356237309515f
#define SQRTPIF  1.77245385090551603f
#define CG_ITERS 256

// gfx1250 has a native v_tanh_f32 transcendental; use it when the builtin is
// declared, otherwise fall back to the OCML expansion.
__device__ __forceinline__ float fast_tanh(float x)
{
#if __has_builtin(__builtin_amdgcn_tanhf)
    return __builtin_amdgcn_tanhf(x);
#else
    return tanhf(x);
#endif
}

// ---------------------------------------------------------------------------
// Kernel 1: chi = tanhshrink(tanhshrink(tanhshrink(X@W1+b1)@W2+b2)@W3+b3)
// One wave per 16-row tile of X. f32 -> f16 operands, f32 accumulation via
// v_wmma_f32_16x16x32_f16. Fragment layouts per CDNA5 ISA 7.12.2 (wave32):
//   A 16x32 f16 : lane<16 row=lane  holds K = kc*32 + {0..7, 16..23}
//                 lane>=16 row=lane-16 holds K = kc*32 + {8..15, 24..31}
//   B 32x16 f16 : lane<16 col=lane  holds K = kc*32 + {0..7, 16..23}
//                 lane>=16 col=lane-16 holds K = kc*32 + {8..15, 24..31}
//   C/D 16x16 f32: lane<16 N=lane M=v ; lane>=16 N=lane-16 M=v+8
// ---------------------------------------------------------------------------
__global__ __launch_bounds__(256) void chi_mlp_kernel(
    const float* __restrict__ X,  const float* __restrict__ W1, const float* __restrict__ b1,
    const float* __restrict__ W2, const float* __restrict__ b2,
    const float* __restrict__ W3, const float* __restrict__ b3,
    float* __restrict__ chi)
{
    __shared__ float lds[8 * 16 * U_];              // 4KB per wave, 32KB per block
    const int lane = threadIdx.x & 31;
    const int wave = threadIdx.x >> 5;
    const int tile = blockIdx.x * 8 + wave;         // 16-row tile of (B*N) rows
    const int row0 = tile * 16;
    float*     scratch   = lds + wave * 16 * U_;    // wave-private
    _Float16*  scratch_h = (_Float16*)scratch;
    const int  m  = lane & 15;
    const bool hi = lane >= 16;

    // ---------------- layer 1: 16x128 @ 128x64 ----------------
    v16h a1[4];
    {
        const float* xr = X + (size_t)(row0 + m) * D_;
        #pragma unroll
        for (int kc = 0; kc < 4; ++kc) {
            const int kb = kc * 32 + (hi ? 8 : 0);
            v16h a;
            #pragma unroll
            for (int i = 0; i < 8; ++i) {
                a[i]     = (_Float16)xr[kb + i];
                a[i + 8] = (_Float16)xr[kb + 16 + i];
            }
            a1[kc] = a;
        }
    }
    #pragma unroll
    for (int nt = 0; nt < 4; ++nt) {
        const int col = nt * 16 + m;
        v8f c = {};
        #pragma unroll
        for (int kc = 0; kc < 4; ++kc) {
            const int kb = kc * 32 + (hi ? 8 : 0);
            v16h bf;
            #pragma unroll
            for (int j = 0; j < 16; ++j)
                bf[j] = (_Float16)W1[(kb + j + (j >= 8 ? 8 : 0)) * U_ + col];
            c = __builtin_amdgcn_wmma_f32_16x16x32_f16(false, a1[kc], false, bf,
                                                       (short)0, c, false, false);
        }
        const float bias = b1[col];
        const int   mb   = hi ? 8 : 0;
        #pragma unroll
        for (int v = 0; v < 8; ++v) {
            float x = c[v] + bias;
            scratch_h[(mb + v) * U_ + col] = (_Float16)(x - fast_tanh(x));
        }
    }

    // ---------------- layer 2: 16x64 @ 64x64 ----------------
    v16h a2[2];
    #pragma unroll
    for (int kc = 0; kc < 2; ++kc) {
        const int kb = kc * 32 + (hi ? 8 : 0);
        v16h a;
        #pragma unroll
        for (int i = 0; i < 8; ++i) {
            a[i]     = scratch_h[m * U_ + kb + i];
            a[i + 8] = scratch_h[m * U_ + kb + 16 + i];
        }
        a2[kc] = a;
    }
    #pragma unroll
    for (int nt = 0; nt < 4; ++nt) {
        const int col = nt * 16 + m;
        v8f c = {};
        #pragma unroll
        for (int kc = 0; kc < 2; ++kc) {
            const int kb = kc * 32 + (hi ? 8 : 0);
            v16h bf;
            #pragma unroll
            for (int j = 0; j < 16; ++j)
                bf[j] = (_Float16)W2[(kb + j + (j >= 8 ? 8 : 0)) * U_ + col];
            c = __builtin_amdgcn_wmma_f32_16x16x32_f16(false, a2[kc], false, bf,
                                                       (short)0, c, false, false);
        }
        const float bias = b2[col];
        const int   mb   = hi ? 8 : 0;
        #pragma unroll
        for (int v = 0; v < 8; ++v) {
            float x = c[v] + bias;
            scratch[(mb + v) * U_ + col] = x - fast_tanh(x); // f32 overwrite; a2 in regs
        }
    }

    // ---------------- layer 3: 16x64 @ 64x1 (matvec, VALU) ----------------
    if (lane < 16) {
        float acc = b3[0];
        #pragma unroll 8
        for (int j = 0; j < U_; ++j) acc += scratch[lane * U_ + j] * W3[j];
        chi[row0 + lane] = acc - fast_tanh(acc);
    }
}

// ---------------------------------------------------------------------------
// Kernel 2: A[b][i][j] = k*erf(d_ij/(sqrt2*gamma_ij))/d_ij  (+ diag terms)
// One block per 64-row slab; pos/sigma/hardness of the batch cached in LDS.
// float4 streaming stores: 67MB of writes ~= 3us @ 23.3 TB/s.
// ---------------------------------------------------------------------------
__global__ __launch_bounds__(256) void build_A_kernel(
    const float* __restrict__ pos, const int* __restrict__ ntype,
    const float* __restrict__ hardness, const float* __restrict__ sigma,
    float* __restrict__ Aout)
{
    __shared__ float spx[N_], spy[N_], spz[N_], ssig2[N_], sdiag[N_];
    const int b    = blockIdx.x >> 3;    // 8 row-slabs per batch
    const int rblk = blockIdx.x & 7;
    for (int i = threadIdx.x; i < N_; i += 256) {
        const size_t gi = (size_t)b * N_ + i;
        spx[i] = pos[gi * 3 + 0];
        spy[i] = pos[gi * 3 + 1];
        spz[i] = pos[gi * 3 + 2];
        const int   t = ntype[gi];
        const float s = sigma[t];
        ssig2[i] = s * s;
        sdiag[i] = hardness[t] + KCOUL / (SQRTPIF * SQRT2F * s);
    }
    __syncthreads();
    float4* out4 = (float4*)(Aout + (size_t)b * N_ * N_ + (size_t)rblk * 64 * N_);
    for (int i4 = threadIdx.x; i4 < (64 * N_) / 4; i4 += 256) {
        const int r  = i4 >> 7;            // row within slab (128 float4 per row)
        const int c0 = (i4 & 127) * 4;     // first column of the quad
        const int gi = rblk * 64 + r;      // global row
        const float px = spx[gi], py = spy[gi], pz = spz[gi];
        const float s2 = ssig2[gi];
        float res[4];
        #pragma unroll
        for (int u = 0; u < 4; ++u) {
            const int cc = c0 + u;
            const float dx = px - spx[cc];
            const float dy = py - spy[cc];
            const float dz = pz - spz[cc];
            const float dist  = sqrtf(dx * dx + dy * dy + dz * dz) + 1e-8f;
            const float gamma = sqrtf(s2 + ssig2[cc]);
            float a = KCOUL * erff(dist / (SQRT2F * gamma)) / dist;
            if (gi == cc) a += sdiag[gi];
            res[u] = a;
        }
        out4[i4] = make_float4(res[0], res[1], res[2], res[3]);  // b128 store
    }
}

// ---------------------------------------------------------------------------
// Kernel 3: bordered solve via Schur complement + Jacobi-PCG (2 RHS share A).
// One block (512 threads = 16 waves) per batch; A (1MB/batch, 64MB total)
// stays L2-resident. Matvec: one row per wave, per-lane float4 (b128 loads),
// global_prefetch of the next row.
// ---------------------------------------------------------------------------
__device__ __forceinline__ float block_reduce_512(float v, float* sred, int tid)
{
    #pragma unroll
    for (int o = 16; o > 0; o >>= 1) v += __shfl_down(v, o, 32);
    if ((tid & 31) == 0) sred[tid >> 5] = v;
    __syncthreads();
    if (tid < 32) {
        float s = (tid < 16) ? sred[tid] : 0.0f;
        #pragma unroll
        for (int o = 8; o > 0; o >>= 1) s += __shfl_down(s, o, 32);
        if (tid == 0) sred[0] = s;
    }
    __syncthreads();
    const float r = sred[0];
    __syncthreads();
    return r;
}

__global__ __launch_bounds__(512) void qeq_solve_kernel(
    const float* __restrict__ Amat, const float* __restrict__ chi,
    const float* __restrict__ total_charge, float* __restrict__ charges)
{
    __shared__ __align__(16) float py[N_];
    __shared__ __align__(16) float pz[N_];
    __shared__ float sApy[N_], sApz[N_], sred[16];
    const int b    = blockIdx.x;
    const int tid  = threadIdx.x;
    const int lane = tid & 31;
    const int wave = tid >> 5;
    const float* Ab = Amat + (size_t)b * N_ * N_;

    // A y = chi ; A z = 1   (A is SPD). Jacobi preconditioner from diag(A).
    const float dinv = 1.0f / Ab[(size_t)tid * N_ + tid];
    float ry = chi[(size_t)b * N_ + tid];
    float rz = 1.0f;
    float xy = 0.0f, xz = 0.0f;
    float zy = dinv * ry, zz = dinv * rz;
    py[tid] = zy; pz[tid] = zz;
    float rhoy = block_reduce_512(ry * zy, sred, tid);
    float rhoz = block_reduce_512(rz * zz, sred, tid);

    const float4* py4 = (const float4*)py;
    const float4* pz4 = (const float4*)pz;

    for (int it = 0; it < CG_ITERS; ++it) {
        // Ap = A*p for both RHS: each A element loaded once (b128), used twice.
        for (int rr = 0; rr < 32; ++rr) {
            const int row = wave * 32 + rr;
            const float4* Arow4 = (const float4*)(Ab + (size_t)row * N_);
            if (rr + 1 < 32)
                __builtin_prefetch(Ab + (size_t)(row + 1) * N_ + lane * 16, 0, 1);
            float ay = 0.0f, az = 0.0f;
            #pragma unroll
            for (int jj = 0; jj < 4; ++jj) {
                const int j4 = jj * 32 + lane;      // coalesced 512B/wave b128 loads
                const float4 a4 = Arow4[j4];
                const float4 p4 = py4[j4];
                const float4 q4 = pz4[j4];
                ay += a4.x * p4.x + a4.y * p4.y + a4.z * p4.z + a4.w * p4.w;
                az += a4.x * q4.x + a4.y * q4.y + a4.z * q4.z + a4.w * q4.w;
            }
            #pragma unroll
            for (int o = 16; o > 0; o >>= 1) {
                ay += __shfl_down(ay, o, 32);
                az += __shfl_down(az, o, 32);
            }
            if (lane == 0) { sApy[row] = ay; sApz[row] = az; }
        }
        __syncthreads();
        const float Apy = sApy[tid], Apz = sApz[tid];
        const float pApy = block_reduce_512(py[tid] * Apy, sred, tid);
        const float pApz = block_reduce_512(pz[tid] * Apz, sred, tid);
        const float aly = rhoy / pApy;
        const float alz = rhoz / pApz;
        xy += aly * py[tid];  xz += alz * pz[tid];
        ry -= aly * Apy;      rz -= alz * Apz;
        zy = dinv * ry;       zz = dinv * rz;
        const float rny = block_reduce_512(ry * zy, sred, tid);
        const float rnz = block_reduce_512(rz * zz, sred, tid);
        const float bty = rny / rhoy;
        const float btz = rnz / rhoz;
        rhoy = rny; rhoz = rnz;
        py[tid] = zy + bty * py[tid];
        pz[tid] = zz + btz * pz[tid];
        __syncthreads();
    }

    // lambda = -(Q + 1'y)/(1'z) ; q = -(y + lambda z)
    const float Sy = block_reduce_512(xy, sred, tid);
    const float Sz = block_reduce_512(xz, sred, tid);
    const float lambda = -(total_charge[b] + Sy) / Sz;
    charges[(size_t)b * N_ + tid] = -(xy + lambda * xz);
}

// ---------------------------------------------------------------------------
extern "C" void kernel_launch(void* const* d_in, const int* in_sizes, int n_in,
                              void* d_out, int out_size, void* d_ws, size_t ws_size,
                              hipStream_t stream)
{
    const float* pos        = (const float*)d_in[0];
    const float* node_feats = (const float*)d_in[1];
    const int*   node_type  = (const int*)  d_in[2];
    const float* total_q    = (const float*)d_in[3];
    const float* hardness   = (const float*)d_in[4];
    const float* sigma      = (const float*)d_in[5];
    const float* W1 = (const float*)d_in[6];
    const float* b1 = (const float*)d_in[7];
    const float* W2 = (const float*)d_in[8];
    const float* b2 = (const float*)d_in[9];
    const float* W3 = (const float*)d_in[10];
    const float* b3 = (const float*)d_in[11];

    float* charges = (float*)d_out;                       // (B,N)
    float* Aout    = (float*)d_out + (size_t)B_ * N_;     // (B,N,N) — second output
    float* chiWs   = (float*)d_ws;                        // (B,N) scratch

    chi_mlp_kernel<<<(B_ * N_) / (16 * 8), 256, 0, stream>>>(
        node_feats, W1, b1, W2, b2, W3, b3, chiWs);
    build_A_kernel<<<B_ * (N_ / 64), 256, 0, stream>>>(
        pos, node_type, hardness, sigma, Aout);
    qeq_solve_kernel<<<B_, 512, 0, stream>>>(
        Aout, chiWs, total_q, charges);

    (void)in_sizes; (void)n_in; (void)out_size; (void)ws_size;
}
